// WeightGRU_77232101917081
// MI455X (gfx1250) — matrix operand order repaired and checked by reading, not compiled, not run on hardware
//
#include <hip/hip_runtime.h>
#include <hip/hip_bf16.h>

typedef __attribute__((ext_vector_type(8)))  float  v8f;
typedef __attribute__((ext_vector_type(16))) __bf16 v16bf;
typedef __attribute__((ext_vector_type(8)))  __bf16 v8bf;

#define GRU_T   16384
#define GRU_H   512
#define GRU_G3  1536
#define SCAN_NBLK 16

// ---------------------------------------------------------------------------
// f32 -> bf16 conversion (grid-stride)
// ---------------------------------------------------------------------------
__global__ void cvt_bf16_kernel(const float* __restrict__ s, __bf16* __restrict__ d, int n) {
    int i = blockIdx.x * blockDim.x + threadIdx.x;
    int stride = gridDim.x * blockDim.x;
    for (; i < n; i += stride) d[i] = (__bf16)s[i];
}

// ---------------------------------------------------------------------------
// WMMA GEMM: C[M,N] = A[M,K](bf16 rm) * B[N,K](bf16 rm)^T + bias[N]
//
// block = 256 threads = 8 waves arranged 2(M) x 4(N); block tile 64M x 128N.
// Each wave computes a 32x32 register tile = 2x2 WMMA tiles (A/B fragments
// reused twice each -> 4 WMMAs per 2A+2B fragment loads).
//
// A k-slices (64 rows x 32 K, 4KB) are double-buffered into LDS with
// global_load_async_to_lds_b128 (one b128 per thread), synchronized with
// s_wait_asynccnt; rows padded to 40 bf16 (20 dwords) for conflict-free
// ds_load_b128 fragment reads.
//
// Per-lane WMMA layouts used here (A/B use contiguous k, not the ISA table order):
//   A 16x32 bf16: lane l, m=l&15, K=(l>>4)*16+{0..15}
//   B 32x16 bf16: n=l&15, K=(l>>4)*16+{0..15}  (16 contiguous bf16 of a [N,K] row)
//   C/D f32:      row=(l>>4)*8+i, col=l&15
// ---------------------------------------------------------------------------
__global__ __launch_bounds__(256)
void wmma_gemm_bias_kernel(const __bf16* __restrict__ A, const __bf16* __restrict__ B,
                           const float* __restrict__ bias,
                           float* __restrict__ Cf, __bf16* __restrict__ Cbf,
                           int M, int N, int K) {
    constexpr int APAD = 40;                       // padded LDS row stride (elems)
    __shared__ __align__(16) __bf16 abuf[2][64 * APAD];

    const int tid   = threadIdx.x;
    const int lane  = tid & 31;
    const int wave  = tid >> 5;
    const int waveM = wave >> 2;                   // 0..1
    const int waveN = wave & 3;                    // 0..3
    const int mblk  = blockIdx.y * 64;
    const int nblk  = blockIdx.x * 128;

    const int half = lane >> 4;                    // 0 or 1
    const int r16  = lane & 15;

    // async A staging: thread i copies one 16B chunk: row = i>>2, chunk = i&3
    const int frow = tid >> 2, fchk = tid & 3;
    const __bf16* fsrc0 = A + (size_t)(mblk + frow) * K + fchk * 8;
    const unsigned fdst0 = (unsigned)(size_t)&abuf[0][frow * APAD + fchk * 8];
    const unsigned fdst1 = (unsigned)(size_t)&abuf[1][frow * APAD + fchk * 8];

    // B fragment base pointers (direct from global, L0/L2-resident)
    const __bf16* brow0 = B + (size_t)(nblk + waveN * 32 + r16) * K + half * 16;
    const __bf16* brow1 = brow0 + (size_t)16 * K;

    // A fragment base pointer within LDS buffer 0
    const __bf16* arow0 = &abuf[0][0] + (waveM * 32 + r16) * APAD + half * 16;

    v8f acc00 = {}, acc01 = {}, acc10 = {}, acc11 = {};

    const int KT = K / 32;
    // prime buffer 0
    asm volatile("global_load_async_to_lds_b128 %0, %1, off"
                 :: "v"(fdst0), "v"(fsrc0) : "memory");

    for (int kt = 0; kt < KT; ++kt) {
        const int cur = kt & 1;
        if (kt + 1 < KT) {                          // issue fill of the other buffer
            const __bf16* fsrc = fsrc0 + (size_t)(kt + 1) * 32;
            const unsigned fdst = cur ? fdst0 : fdst1;
            asm volatile("global_load_async_to_lds_b128 %0, %1, off"
                         :: "v"(fdst), "v"(fsrc) : "memory");
            asm volatile("s_wait_asynccnt 0x1" ::: "memory");   // oldest fill done
        } else {
            asm volatile("s_wait_asynccnt 0x0" ::: "memory");
        }
        __syncthreads();                            // all threads' copies visible

        const __bf16* ab = arow0 + cur * (64 * APAD);
        v8bf a00 = *(const v8bf*)(ab);              // msub=0, K base..base+7
        v8bf a01 = *(const v8bf*)(ab + 8);          // msub=0, K base+8..base+15
        v8bf a10 = *(const v8bf*)(ab + 16 * APAD);  // msub=1
        v8bf a11 = *(const v8bf*)(ab + 16 * APAD + 8);
        v16bf am0, am1;
#pragma unroll
        for (int i = 0; i < 8; ++i) {
            am0[i] = a00[i]; am0[i + 8] = a01[i];
            am1[i] = a10[i]; am1[i + 8] = a11[i];
        }
        const int k0 = kt * 32;
        v16bf bm0 = *(const v16bf*)(brow0 + k0);
        v16bf bm1 = *(const v16bf*)(brow1 + k0);
        __builtin_prefetch(brow0 + k0 + 64, 0, 1);
        __builtin_prefetch(brow1 + k0 + 64, 0, 1);

        acc00 = __builtin_amdgcn_wmma_f32_16x16x32_bf16(false, am0, false, bm0, (short)0, acc00, false, false);
        acc01 = __builtin_amdgcn_wmma_f32_16x16x32_bf16(false, am0, false, bm1, (short)0, acc01, false, false);
        acc10 = __builtin_amdgcn_wmma_f32_16x16x32_bf16(false, am1, false, bm0, (short)0, acc10, false, false);
        acc11 = __builtin_amdgcn_wmma_f32_16x16x32_bf16(false, am1, false, bm1, (short)0, acc11, false, false);
        __syncthreads();                            // buffer may be overwritten next iter
    }

    auto store_tile = [&](const v8f& acc, int m0, int n0) {
        const int col = n0 + r16;
        const float bv = bias ? bias[col] : 0.0f;
#pragma unroll
        for (int i = 0; i < 8; ++i) {
            const int row = m0 + half * 8 + i;
            const float v = acc[i] + bv;
            if (Cf)  Cf[(size_t)row * N + col] = v;
            if (Cbf) Cbf[(size_t)row * N + col] = (__bf16)v;
        }
    };
    const int mw = mblk + waveM * 32;
    const int nw = nblk + waveN * 32;
    store_tile(acc00, mw,      nw);
    store_tile(acc01, mw,      nw + 16);
    store_tile(acc10, mw + 16, nw);
    store_tile(acc11, mw + 16, nw + 16);
}

// ---------------------------------------------------------------------------
// init: h = 0, global barrier counter = 0 (must run every call: ws is not
// re-poisoned between graph replays)
// ---------------------------------------------------------------------------
__global__ void init_state_kernel(float* __restrict__ h, unsigned* __restrict__ ctr) {
    int i = threadIdx.x;
    if (i < GRU_H) h[i] = 0.0f;
    if (i == 0) *ctr = 0u;
}

// ---------------------------------------------------------------------------
// Sequential GRU scan. 16 persistent blocks x 256 threads. Block b owns
// h-indices [b*32, b*32+32) i.e. w_hh rows {g*512 + j}. Per step:
//   - every lane holds h[lane*16 .. lane*16+15] in regs (2KB, L2-hot)
//   - each wave computes 12 row-dots (16 bf16 FMAs/lane + 5 shfl_xor)
//   - threads 0..31 apply gate nonlinearity, update h, emit hs (bf16)
//   - monotonic global atomic barrier (counter reaches 16*(t+1))
// ---------------------------------------------------------------------------
__global__ __launch_bounds__(256)
void gru_scan_kernel(const float* __restrict__ igates,   // [T, 3H] (ir|iz|in), bias folded in
                     const __bf16* __restrict__ whh,     // [3H, H] row-major bf16
                     const float* __restrict__ b_n,      // [H]
                     float* __restrict__ h,              // [H] global state
                     __bf16* __restrict__ hs,            // [T, H] bf16 for decoder GEMM
                     unsigned* __restrict__ ctr, int T) {
    __shared__ float gdots[96];
    const int tid  = threadIdx.x;
    const int lane = tid & 31;
    const int wave = tid >> 5;
    const int jbase = blockIdx.x * 32;

    for (int t = 0; t < T; ++t) {
        // current h into registers (all waves read the same hot 2KB)
        float hreg[16];
#pragma unroll
        for (int i = 0; i < 4; ++i) {
            float4 v4 = *(const float4*)(h + lane * 16 + i * 4);
            hreg[i * 4 + 0] = v4.x; hreg[i * 4 + 1] = v4.y;
            hreg[i * 4 + 2] = v4.z; hreg[i * 4 + 3] = v4.w;
        }
        // 12 rows per wave (8 waves x 12 = 96 rows = 3 gates x 32 h-indices)
#pragma unroll 4
        for (int rr0 = 0; rr0 < 12; ++rr0) {
            const int rr = wave * 12 + rr0;
            const int g = rr >> 5, jj = rr & 31;
            const __bf16* wrow = whh + (size_t)(g * GRU_H + jbase + jj) * GRU_H + lane * 16;
            v8bf w0 = *(const v8bf*)(wrow);
            v8bf w1 = *(const v8bf*)(wrow + 8);
            float s = 0.0f;
#pragma unroll
            for (int i = 0; i < 8; ++i)
                s += (float)w0[i] * hreg[i] + (float)w1[i] * hreg[8 + i];
#pragma unroll
            for (int off = 16; off; off >>= 1) s += __shfl_xor(s, off, 32);
            if (lane == 0) gdots[rr] = s;
        }
        __syncthreads();

        if (tid < 32) {
            const int jj = tid, j = jbase + jj;
            const float* igt = igates + (size_t)t * GRU_G3;
            __builtin_prefetch(igt + GRU_G3 + j, 0, 1);          // next step's gates
            const float hr = gdots[jj], hz = gdots[32 + jj], hn = gdots[64 + jj];
            const float ir  = igt[j];
            const float iz  = igt[GRU_H + j];
            const float inn = igt[2 * GRU_H + j];
            const float hold = h[j];
            const float r = 1.0f / (1.0f + __expf(-(ir + hr)));
            const float z = 1.0f / (1.0f + __expf(-(iz + hz)));
            const float pre = inn + r * (hn + b_n[j]);
            const float n = 2.0f / (1.0f + __expf(-2.0f * pre)) - 1.0f;   // tanh
            const float hnew = n + z * (hold - n);
            h[j] = hnew;
            hs[(size_t)t * GRU_H + j] = (__bf16)hnew;
        }
        __threadfence();
        __syncthreads();
        if (tid == 0) {
            const unsigned target = (unsigned)(t + 1) * SCAN_NBLK;
            atomicAdd(ctr, 1u);
            while (__hip_atomic_load(ctr, __ATOMIC_RELAXED, __HIP_MEMORY_SCOPE_AGENT) < target)
                __builtin_amdgcn_s_sleep(1);
        }
        __syncthreads();
    }
}

// ---------------------------------------------------------------------------
extern "C" void kernel_launch(void* const* d_in, const int* in_sizes, int n_in,
                              void* d_out, int out_size, void* d_ws, size_t ws_size,
                              hipStream_t stream) {
    (void)in_sizes; (void)n_in; (void)out_size; (void)ws_size;
    const int T = GRU_T, IN = GRU_H, H = GRU_H, G3 = GRU_G3;

    const float* inputs = (const float*)d_in[0];   // [T, IN]
    const float* enc_w  = (const float*)d_in[1];   // [H, IN]
    const float* enc_b  = (const float*)d_in[2];   // [H]
    const float* w_ih   = (const float*)d_in[3];   // [3H, H]
    const float* w_hh   = (const float*)d_in[4];   // [3H, H]
    const float* b_ih   = (const float*)d_in[5];   // [3H]
    const float* b_n    = (const float*)d_in[6];   // [H]
    const float* dec_w  = (const float*)d_in[7];   // [IN, H]
    const float* dec_b  = (const float*)d_in[8];   // [IN]
    float* preds = (float*)d_out;                  // [T, IN]

    char* ws = (char*)d_ws;
    size_t o = 0;
    __bf16* x_bf    = (__bf16*)(ws + o); o += (size_t)T * IN * 2;
    __bf16* encw_bf = (__bf16*)(ws + o); o += (size_t)H * IN * 2;
    __bf16* wih_bf  = (__bf16*)(ws + o); o += (size_t)G3 * H * 2;
    __bf16* whh_bf  = (__bf16*)(ws + o); o += (size_t)G3 * H * 2;
    __bf16* decw_bf = (__bf16*)(ws + o); o += (size_t)IN * H * 2;
    __bf16* emb_bf  = (__bf16*)(ws + o); o += (size_t)T * H * 2;
    __bf16* hs_bf   = (__bf16*)(ws + o); o += (size_t)T * H * 2;
    float*  igates  = (float*)(ws + o);  o += (size_t)T * G3 * 4;
    float*  hbuf    = (float*)(ws + o);  o += (size_t)H * 4;
    unsigned* ctr   = (unsigned*)(ws + o);

    // --- precision conversion (bf16 operands, f32 accumulate) ---
    cvt_bf16_kernel<<<1024, 256, 0, stream>>>(inputs, x_bf,    T * IN);
    cvt_bf16_kernel<<<256,  256, 0, stream>>>(enc_w,  encw_bf, H * IN);
    cvt_bf16_kernel<<<512,  256, 0, stream>>>(w_ih,   wih_bf,  G3 * H);
    cvt_bf16_kernel<<<512,  256, 0, stream>>>(w_hh,   whh_bf,  G3 * H);
    cvt_bf16_kernel<<<256,  256, 0, stream>>>(dec_w,  decw_bf, IN * H);

    // --- emb = X @ enc_w^T + enc_b  (bf16 out only, feeds next GEMM) ---
    wmma_gemm_bias_kernel<<<dim3(H / 128, T / 64), 256, 0, stream>>>(
        x_bf, encw_bf, enc_b, nullptr, emb_bf, T, H, IN);

    // --- igates = emb @ w_ih^T + b_ih  (f32, consumed by the scan) ---
    wmma_gemm_bias_kernel<<<dim3(G3 / 128, T / 64), 256, 0, stream>>>(
        emb_bf, wih_bf, b_ih, igates, nullptr, T, G3, H);

    // --- sequential GRU scan across 16 persistent blocks ---
    init_state_kernel<<<1, 512, 0, stream>>>(hbuf, ctr);
    gru_scan_kernel<<<SCAN_NBLK, 256, 0, stream>>>(igates, whh_bf, b_n, hbuf, hs_bf, ctr, T);

    // --- preds = hs @ dec_w^T + dec_b ---
    wmma_gemm_bias_kernel<<<dim3(IN / 128, T / 64), 256, 0, stream>>>(
        hs_bf, decw_bf, dec_b, preds, nullptr, T, IN, H);
}
